// SentimentClassifier_51625506898358
// MI455X (gfx1250) — matrix operand order repaired and checked by reading, not compiled, not run on hardware
//
#include <hip/hip_runtime.h>
#include <hip/hip_bf16.h>

#define BATCH 64
#define SEQ   512
#define EMBD  512
#define HID   512
#define GATES 2048   /* 4*HID */
#define KDIM  1024   /* EMB+HID or HID+HID */
#define NBLK  64
#define NTHR  256

typedef __attribute__((ext_vector_type(16))) _Float16 v16h;
typedef __attribute__((ext_vector_type(8)))  _Float16 v8h;
typedef __attribute__((ext_vector_type(8)))  float    v8f;
typedef __attribute__((ext_vector_type(4)))  float    v4f;

// ---------------- stage A panel (16 rows x 1024 K) into LDS ---------------
// 256 threads: row = tid>>4 (0..15), seg = tid&15 -> 64 halves each.

__device__ __forceinline__ void stage_emb_h(
    _Float16* __restrict__ As, const int* __restrict__ x,
    const float* __restrict__ emb, const _Float16* __restrict__ hsrc,
    int m0, int t, int srow, int scol) {
  _Float16* dst = As + srow * KDIM + scol;
  if (scol < 512) {                       // embedding half (f32 -> f16)
    int token = x[(m0 + srow) * SEQ + t];
    if (token == 0) {                     // padding_idx = 0
      v8h z = {};
#pragma unroll
      for (int j = 0; j < 8; ++j) ((v8h*)dst)[j] = z;
    } else {
      const float* src = emb + (long)token * EMBD + scol;
#pragma unroll
      for (int j = 0; j < 8; ++j) {
        v4f f0 = ((const v4f*)src)[2 * j];
        v4f f1 = ((const v4f*)src)[2 * j + 1];
        v8h h8;
#pragma unroll
        for (int i = 0; i < 4; ++i) {
          h8[i]     = (_Float16)f0[i];
          h8[4 + i] = (_Float16)f1[i];
        }
        ((v8h*)dst)[j] = h8;
      }
    }
  } else {                                // hidden-state half (f16 copy)
    const _Float16* src = hsrc + (m0 + srow) * HID + (scol - 512);
#pragma unroll
    for (int j = 0; j < 8; ++j) ((v8h*)dst)[j] = ((const v8h*)src)[j];
  }
}

__device__ __forceinline__ void stage_hh(
    _Float16* __restrict__ As, const _Float16* __restrict__ h0,
    const _Float16* __restrict__ h1, int m0, int srow, int scol) {
  _Float16* dst = As + srow * KDIM + scol;
  const _Float16* src = (scol < 512)
                            ? (h0 + (m0 + srow) * HID + scol)
                            : (h1 + (m0 + srow) * HID + (scol - 512));
#pragma unroll
  for (int j = 0; j < 8; ++j) ((v8h*)dst)[j] = ((const v8h*)src)[j];
}

// ---------------- one 16x16 tile over K=1024, A from LDS -------------------
// A operand k order used here (contiguous, as in packed B): lanes 0-15 rows, elems 0-15 = K kA..kA+15,
// kA = kb*32 + (lane>>4)*16.
__device__ __forceinline__ void gemm_tile_lds(
    const _Float16* __restrict__ As, const _Float16* __restrict__ wp,
    float* __restrict__ gates, int m0, int nt, int lane) {
  v8f acc = {};
  const int arow   = lane & 15;
  const int kA_off = (lane >> 4) * 16;
#pragma unroll 4
  for (int kb = 0; kb < 32; ++kb) {
    const v8h* p = (const v8h*)(As + arow * KDIM + kb * 32 + kA_off);
    v8h a0 = p[0];
    v8h a1 = p[1];                        // +8 halves
    v16h a;
#pragma unroll
    for (int i = 0; i < 8; ++i) { a[i] = a0[i]; a[8 + i] = a1[i]; }
    v16h b = *(const v16h*)(wp + ((kb * 128 + nt) * 32 + lane) * 16);
    acc = __builtin_amdgcn_wmma_f32_16x16x32_f16(false, a, false, b,
                                                 (short)0, acc, false, false);
  }
  // C/D layout: VGPR r -> row m0+r (lanes 0-15) / m0+r+8 (lanes 16-31)
  const int n = nt * 16 + (lane & 15);
#pragma unroll
  for (int r = 0; r < 8; ++r)
    gates[(m0 + r + 8 * (lane >> 4)) * GATES + n] = acc[r];
}

// ---------------- device-wide barrier (arrive + monotonic target) ----------
__device__ __forceinline__ void grid_barrier(unsigned* bar, unsigned gen) {
  __syncthreads();
  if (threadIdx.x == 0) {
    __hip_atomic_fetch_add(bar, 1u, __ATOMIC_ACQ_REL, __HIP_MEMORY_SCOPE_AGENT);
    const unsigned target = gen * NBLK;
    while (__hip_atomic_load(bar, __ATOMIC_ACQUIRE, __HIP_MEMORY_SCOPE_AGENT) <
           target)
      __builtin_amdgcn_s_sleep(1);
  }
  __syncthreads();
}

__device__ __forceinline__ float sigf(float x) {
  return 1.0f / (1.0f + __expf(-x));
}

// gates(+bias) -> i,f,o,g -> c,h update (relu variant per reference)
__device__ __forceinline__ void cell_update(const float* __restrict__ gates,
                                            const float* __restrict__ bias,
                                            _Float16* __restrict__ h,
                                            float* __restrict__ c, int tid) {
#pragma unroll
  for (int it = 0; it < 2; ++it) {
    int idx = tid + it * (NBLK * NTHR);   // 0..32767 over B*HID
    int bb = idx >> 9, hh = idx & 511;
    const float* g = gates + bb * GATES;
    float i_ = sigf(g[hh]         + bias[hh]);
    float f_ = sigf(g[512 + hh]   + bias[512 + hh]);
    float o_ = sigf(g[1024 + hh]  + bias[1024 + hh]);
    float gg = fmaxf(g[1536 + hh] + bias[1536 + hh], 0.0f);
    float cn = f_ * c[idx] + i_ * gg;
    c[idx] = cn;
    h[idx] = (_Float16)(o_ * fmaxf(cn, 0.0f));
  }
}

// ---------------- kernels --------------------------------------------------
__global__ void init_state(_Float16* h0, _Float16* h1, float* c0, float* c1,
                           unsigned* bar) {
  int i = blockIdx.x * blockDim.x + threadIdx.x;
  if (i < BATCH * HID) {
    h0[i] = (_Float16)0.0f; h1[i] = (_Float16)0.0f;
    c0[i] = 0.0f;           c1[i] = 0.0f;
  }
  if (i == 0) *bar = 0u;
}

// Repack W[g][h][k] f32 -> per-lane WMMA B-fragment layout, f16:
//   dst[((kb*128 + nb)*32 + lane)*16 + e] = W[n>>9][n&511][k]
//   with k = kb*32 + (lane>>4)*16 + e, n = nb*16 + (lane&15)
__global__ void pack_weights(const float* __restrict__ W0,
                             const float* __restrict__ W1,
                             _Float16* __restrict__ Wp0,
                             _Float16* __restrict__ Wp1) {
  unsigned tid = blockIdx.x * blockDim.x + threadIdx.x; // 0..2*2^21
  int layer = tid >> 21;
  unsigned o = tid & ((1u << 21) - 1);
  int e    = o & 15;
  int lane = (o >> 4) & 31;
  int nb   = (o >> 9) & 127;
  int kb   = o >> 16;                       // 0..31
  int k = kb * 32 + (lane >> 4) * 16 + e;
  int n = nb * 16 + (lane & 15);
  int g = n >> 9, h = n & 511;
  const float* W = layer ? W1 : W0;
  _Float16*   Wp = layer ? Wp1 : Wp0;
  Wp[o] = (_Float16)W[(g * 512 + h) * 1024 + k];
}

__global__ __launch_bounds__(NTHR) void lstm_persistent(
    const int* __restrict__ x, const float* __restrict__ emb,
    const float* __restrict__ b0, const float* __restrict__ b1,
    const float* __restrict__ fc_w, const float* __restrict__ fc_b,
    const _Float16* __restrict__ Wp0, const _Float16* __restrict__ Wp1,
    _Float16* h0, _Float16* h1, float* c0, float* c1, float* gates,
    unsigned* bar, float* out) {
  __shared__ _Float16 As[16 * KDIM];        // 32 KB A panel (shared by 8 waves)
  const int lane = threadIdx.x & 31;
  const int wid  = blockIdx.x * (NTHR / 32) + (threadIdx.x >> 5); // 0..511
  const int m0   = ((blockIdx.x * (NTHR / 32)) >> 7) * 16; // uniform per block
  const int nt   = wid & 127;               // 128 N-tiles
  const int tid  = blockIdx.x * NTHR + threadIdx.x;
  const int srow = threadIdx.x >> 4;        // staging: row 0..15
  const int scol = (threadIdx.x & 15) * 64; // staging: 64 halves per thread
  unsigned gen = 0;
#pragma unroll 1
  for (int t = 0; t < SEQ; ++t) {
    // layer 0: A = [embed(x_t), h0]
    stage_emb_h(As, x, emb, h0, m0, t, srow, scol);
    __syncthreads();
    gemm_tile_lds(As, Wp0, gates, m0, nt, lane);
    grid_barrier(bar, ++gen);
    cell_update(gates, b0, h0, c0, tid);
    grid_barrier(bar, ++gen);
    // layer 1: A = [h0, h1]
    stage_hh(As, h0, h1, m0, srow, scol);
    __syncthreads();
    gemm_tile_lds(As, Wp1, gates, m0, nt, lane);
    grid_barrier(bar, ++gen);
    cell_update(gates, b1, h1, c1, tid);
    grid_barrier(bar, ++gen);
  }
  // final FC + sigmoid: out[b] = sigmoid(h1[b] . fc_w + fc_b)
  if (blockIdx.x == 0 && threadIdx.x < BATCH) {
    int bb = threadIdx.x;
    float s = fc_b[0];
    for (int k = 0; k < HID; ++k) s += (float)h1[bb * HID + k] * fc_w[k];
    out[bb] = 1.0f / (1.0f + __expf(-s));
  }
}

// ---------------- host entry -----------------------------------------------
extern "C" void kernel_launch(void* const* d_in, const int* in_sizes, int n_in,
                              void* d_out, int out_size, void* d_ws,
                              size_t ws_size, hipStream_t stream) {
  const int*   x    = (const int*)d_in[0];
  const float* emb  = (const float*)d_in[1];
  const float* W0   = (const float*)d_in[2];
  const float* b0   = (const float*)d_in[3];
  const float* W1   = (const float*)d_in[4];
  const float* b1   = (const float*)d_in[5];
  const float* fc_w = (const float*)d_in[6];
  const float* fc_b = (const float*)d_in[7];
  float* out = (float*)d_out;

  char* ws = (char*)d_ws;
  size_t off = 0;
  auto alloc = [&](size_t bytes) -> char* {
    char* p = ws + off;
    off += (bytes + 255) & ~size_t(255);
    return p;
  };
  unsigned* bar = (unsigned*)alloc(256);
  _Float16* Wp0 = (_Float16*)alloc((size_t)KDIM * GATES * 2);
  _Float16* Wp1 = (_Float16*)alloc((size_t)KDIM * GATES * 2);
  _Float16* h0  = (_Float16*)alloc(BATCH * HID * 2);
  _Float16* h1  = (_Float16*)alloc(BATCH * HID * 2);
  float* c0     = (float*)alloc(BATCH * HID * 4);
  float* c1     = (float*)alloc(BATCH * HID * 4);
  float* gates  = (float*)alloc(BATCH * GATES * 4);

  hipLaunchKernelGGL(init_state, dim3(128), dim3(256), 0, stream,
                     h0, h1, c0, c1, bar);
  hipLaunchKernelGGL(pack_weights, dim3((2 * KDIM * GATES) / 256), dim3(256),
                     0, stream, W0, W1, Wp0, Wp1);
  hipLaunchKernelGGL(lstm_persistent, dim3(NBLK), dim3(NTHR), 0, stream,
                     x, emb, b0, b1, fc_w, fc_b, Wp0, Wp1,
                     h0, h1, c0, c1, gates, bar, out);
}
